// MetalQuantizedLinear_678604833218
// MI455X (gfx1250) — matrix operand order repaired and checked by reading, not compiled, not run on hardware
//
#include <hip/hip_runtime.h>
#include <hip/hip_fp16.h>

// ---- CDNA5 vector types ----
typedef __attribute__((ext_vector_type(16))) _Float16     v16h;
typedef __attribute__((ext_vector_type(8)))  float        v8f;
typedef __attribute__((ext_vector_type(2)))  _Float16     h2;
typedef __attribute__((ext_vector_type(4)))  unsigned int u32x4;

#define M_DIM   16
#define K_DIM   8192
#define N_DIM   8192
#define GROUP   128
#define WAVES   8
#define KCHUNK  (K_DIM / WAVES)   // 1024 K per wave
#define NTILES  2                 // 16-column tiles per wave sharing one A fragment

// out[16,8192] f16 = x[16,8192] f16 @ dequant(wp[1024,8192] int32 nibbles, scales[64,8192] f16)
__global__ __launch_bounds__(256) void fp4_linear_wmma_kernel(
    const _Float16* __restrict__ x,
    const int*      __restrict__ wp,
    const _Float16* __restrict__ scales,
    _Float16*       __restrict__ out)
{
    __shared__ float red[WAVES * NTILES * 256];   // 16 KB partial tiles

    const int tid    = threadIdx.x;
    const int wave   = tid >> 5;
    const int lane   = tid & 31;
    const int half16 = lane >> 4;                 // which 16-lane half of the wave
    const int l15    = lane & 15;
    const int n0     = blockIdx.x * (16 * NTILES);// this block's 32 output columns

    const int kstart = wave * KCHUNK;
    const int kend   = kstart + KCHUNK;

    v8f acc[NTILES];
#pragma unroll
    for (int t = 0; t < NTILES; ++t) acc[t] = (v8f){};

    // A-fragment addressing: lane holds row (l15), K-offset half16*8 within each 32-K step
    const _Float16* xrow = x + l15 * K_DIM;
    int col[NTILES];
#pragma unroll
    for (int t = 0; t < NTILES; ++t) col[t] = n0 + 16 * t + l15;

    for (int kg = kstart; kg < kend; kg += GROUP) {
        // one scale per (128-K group, column); shared by all 4 WMMA steps below
        h2 hmul[NTILES], hadd[NTILES];
#pragma unroll
        for (int t = 0; t < NTILES; ++t) {
            const float sf = (float)scales[(kg >> 7) * N_DIM + col[t]];
            const _Float16 hm = (_Float16)(0.5f * sf);     // q-term multiplier
            const _Float16 ha = (_Float16)(-516.0f * sf);  // (1024+q)*0.5s-516s = (q-8)*0.5*s
            hmul[t][0] = hm; hmul[t][1] = hm;
            hadd[t][0] = ha; hadd[t][1] = ha;
        }

        // prefetch next group's packed rows while we chew on this one
        if (kg + GROUP < kend) {
#pragma unroll
            for (int t = 0; t < NTILES; ++t)
                __builtin_prefetch(&wp[((kg + GROUP) >> 3) * N_DIM + col[t]], 0, 1);
        }

#pragma unroll
        for (int kk = 0; kk < GROUP; kk += 32) {
            const int k0 = kg + kk;

            // ---- A tile: 16x32 f16, two aligned b128 loads per lane, shared by NTILES ----
            const _Float16* xp = xrow + k0 + half16 * 8;
            union { v16h v; u32x4 q[2]; } ua;
            ua.q[0] = *(const u32x4*)(xp);
            ua.q[1] = *(const u32x4*)(xp + 16);

            // lane covers K = [kb, kb+8) and [kb+16, kb+24) of its column: 2 packed dwords per tile
            const int kb = k0 + half16 * 8;
            const int kp = kb >> 3;

#pragma unroll
            for (int t = 0; t < NTILES; ++t) {
                const unsigned int w0 = (unsigned int)wp[(kp    ) * N_DIM + col[t]];
                const unsigned int w1 = (unsigned int)wp[(kp + 2) * N_DIM + col[t]];

                union { v16h v; h2 p[8]; } ub;
#pragma unroll
                for (int j = 0; j < 4; ++j) {
                    // nibble-pair -> f16 mantissa trick:
                    // {0x6400|q_e, 0x6400|q_o} = {1024+q_e, 1024+q_o}
                    unsigned int t0 = w0 >> (8 * j);
                    unsigned int b0 = (t0 & 0xFu) | ((t0 & 0xF0u) << 12) | 0x64006400u;
                    h2 h0; __builtin_memcpy(&h0, &b0, 4);
                    ub.p[j] = h0 * hmul[t] + hadd[t];          // -> v_pk_fma_f16

                    unsigned int t1 = w1 >> (8 * j);
                    unsigned int b1 = (t1 & 0xFu) | ((t1 & 0xF0u) << 12) | 0x64006400u;
                    h2 h1; __builtin_memcpy(&h1, &b1, 4);
                    ub.p[4 + j] = h1 * hmul[t] + hadd[t];
                }

                // D = A(16x32 f16) x B(32x16 f16) + C, f32 accumulate
                acc[t] = __builtin_amdgcn_wmma_f32_16x16x32_f16(
                    false, ua.v, false, ub.v, (short)0, acc[t], false, false);
            }
        }
    }

    // ---- cross-wave split-K reduction via LDS ----
#pragma unroll
    for (int t = 0; t < NTILES; ++t) {
#pragma unroll
        for (int i = 0; i < 8; ++i) {
            const int m = i + half16 * 8;                      // C/D fragment layout
            red[(wave * NTILES + t) * 256 + m * 16 + l15] = acc[t][i];
        }
    }
    __syncthreads();

    // 256 threads: each finalizes one (m, n) element of each tile
    const int m = tid >> 4;
    const int n = tid & 15;
#pragma unroll
    for (int t = 0; t < NTILES; ++t) {
        float s = 0.0f;
#pragma unroll
        for (int w = 0; w < WAVES; ++w) s += red[(w * NTILES + t) * 256 + tid];
        out[m * N_DIM + n0 + 16 * t + n] = (_Float16)s;
    }
}

extern "C" void kernel_launch(void* const* d_in, const int* in_sizes, int n_in,
                              void* d_out, int out_size, void* d_ws, size_t ws_size,
                              hipStream_t stream) {
    (void)in_sizes; (void)n_in; (void)out_size; (void)d_ws; (void)ws_size;
    const _Float16* x      = (const _Float16*)d_in[0];
    const int*      wp     = (const int*)     d_in[1];
    const _Float16* scales = (const _Float16*)d_in[2];
    _Float16*       out    = (_Float16*)      d_out;

    dim3 grid(N_DIM / (16 * NTILES));  // 256 blocks, one 32-column tile each
    dim3 block(256);                   // 8 wave32 waves splitting K
    hipLaunchKernelGGL(fp4_linear_wmma_kernel, grid, block, 0, stream,
                       x, wp, scales, out);
}
